// SelfAttentionIEU_25555055411260
// MI455X (gfx1250) — matrix operand order repaired and checked by reading, not compiled, not run on hardware
//
#include <hip/hip_runtime.h>
#include <hip/hip_bf16.h>

typedef __attribute__((ext_vector_type(16))) _Float16 v16h;
typedef __attribute__((ext_vector_type(8)))  float    v8f;
typedef __attribute__((ext_vector_type(4)))  unsigned int u32x4;

// Problem constants (from reference): B=16, C=EMBED=256, H=128, W=128, d=20 (pad->32)
#define EMBED   256
#define DPAD    32
#define DREAL   20
#define HDIM    128
#define WDIM    128

// ---------------- LDS layout (~139 KB; 2 blocks / 320KB WGP) ----------------
struct __align__(16) Smem {
    union {
        _Float16 xs[128][272];     // X tile [w][c] f16, padded stride (69632 B)
        float    logits[128][132]; // overlaid after X is dead       (67584 B)
    } ov;
    _Float16 qs[128][40];          // Q [w][d]  (also reused as ctx) (10240 B)
    _Float16 ks[128][40];          // K [w][d]                       (10240 B)
    _Float16 vt[32][136];          // V^T [d][w]                     ( 8704 B)
    _Float16 probs[128][136];      // softmax(QK^T) f16              (34816 B)
    float    outb[16][132];        // per-c-tile output staging      ( 8448 B)
};

// Two 16-byte loads -> one v16h fragment
__device__ __forceinline__ v16h frag2(const _Float16* p0, const _Float16* p1) {
    union { u32x4 u[2]; v16h h; } t;
    t.u[0] = *(const u32x4*)p0;
    t.u[1] = *(const u32x4*)p1;
    return t.h;
}

// A-fragment (16x32 f16) from [M][K] storage, row stride ls (halves).
// ISA layout: lane L -> M = L%16 ; halves: VGPR0-3 = K[hs*8 .. hs*8+7],
// VGPR4-7 = K[16+hs*8 .. 16+hs*8+7], hs = L/16.  Both runs are 16B contiguous.
__device__ __forceinline__ v16h load_a(const _Float16* base, int ls, int mrow,
                                       int kbase, int lane) {
    const _Float16* p = base + (mrow + (lane & 15)) * ls + kbase + ((lane >> 4) << 3);
    return frag2(p, p + 16);
}

// B-fragment (32x16 f16) from [N][K] (transposed) storage, row stride ls.
// lane L -> N = L%16 ; same K order as load_a: K[hs*8..+7], K[16+hs*8..+7], hs = L/16.
__device__ __forceinline__ v16h load_b(const _Float16* base, int ls, int nbase,
                                       int kbase, int lane) {
    const _Float16* p = base + (nbase + (lane & 15)) * ls + kbase + ((lane >> 4) << 3);
    return frag2(p, p + 16);
}

// ------------- prep: f16 transposed zero-padded weights into d_ws -----------
// layout in ws (halves): wqT[32][256] | wkT[32][256] | wvT[32][256] | wpT[256][32]
__global__ void prep_weights(const float* __restrict__ Wq, const float* __restrict__ Wk,
                             const float* __restrict__ Wv, const float* __restrict__ Wp,
                             _Float16* __restrict__ wts) {
    int n = blockIdx.x * blockDim.x + threadIdx.x;
    if (n >= 32 * 256) return;
    _Float16* wqT = wts;
    _Float16* wkT = wts + 32 * 256;
    _Float16* wvT = wts + 2 * 32 * 256;
    _Float16* wpT = wts + 3 * 32 * 256;
    {   // q/k/v: [d][c] <- W[c][d], d padded to 32
        int d = n >> 8, c = n & 255;
        wqT[n] = (_Float16)((d < DREAL) ? Wq[c * DREAL + d] : 0.f);
        wkT[n] = (_Float16)((d < DREAL) ? Wk[c * DREAL + d] : 0.f);
        wvT[n] = (_Float16)((d < DREAL) ? Wv[c * DREAL + d] : 0.f);
    }
    {   // p: [c][d] <- Wp[d][c], d padded to 32
        int c = n >> 5, d = n & 31;
        wpT[n] = (_Float16)((d < DREAL) ? Wp[d * EMBED + c] : 0.f);
    }
}

// ----------------------------- main kernel ----------------------------------
__global__ __launch_bounds__(256)
void attn_row_kernel(const float* __restrict__ x,
                     const float* __restrict__ bq_g, const float* __restrict__ bk_g,
                     const float* __restrict__ bv_g, const float* __restrict__ bp_g,
                     const _Float16* __restrict__ wts,
                     float* __restrict__ out) {
    __shared__ Smem s;
    const int tid  = threadIdx.x;
    const int lane = tid & 31;
    const int wave = tid >> 5;           // 8 waves, wave m owns w-rows [16m,16m+16)
    const int mrow = wave << 4;
    const int bh   = blockIdx.x;
    const int b    = bh >> 7;            // / H
    const int h    = bh & 127;

    const _Float16* wqT = wts;
    const _Float16* wkT = wts + 32 * 256;
    const _Float16* wvT = wts + 2 * 32 * 256;
    const _Float16* wpT = wts + 3 * 32 * 256;
    __builtin_prefetch(wqT, 0, 3);
    __builtin_prefetch(wpT, 0, 3);

    // ---- Phase 1: X[b, :, h, :] (strided [c][w]) -> LDS xs[w][c] as f16 ----
    const float* xb = x + ((size_t)(b * EMBED) * HDIM + h) * WDIM;
    for (int i = tid; i < EMBED * WDIM; i += 256) {
        int c = i >> 7, w = i & 127;                 // consecutive tid = consecutive w
        s.ov.xs[w][c] = (_Float16)xb[(size_t)c * (HDIM * WDIM) + w];
    }
    __syncthreads();

    // ---- Phase 2: Q,K,V = X @ W + b    ([128,256] x [256,32]) ----
    {
        const _Float16* wT[3]   = { wqT, wkT, wvT };
        const float*    bias[3] = { bq_g, bk_g, bv_g };
        for (int mat = 0; mat < 3; ++mat) {
            for (int nt = 0; nt < 2; ++nt) {
                v8f acc = {};
                #pragma unroll
                for (int kst = 0; kst < 8; ++kst) {
                    v16h a = load_a(&s.ov.xs[0][0], 272, mrow, kst * 32, lane);
                    v16h bf = load_b(wT[mat], 256, nt * 16, kst * 32, lane);
                    acc = __builtin_amdgcn_wmma_f32_16x16x32_f16(
                        false, a, false, bf, (short)0, acc, false, false);
                }
                int   col = (nt << 4) | (lane & 15);
                float bb  = (col < DREAL) ? bias[mat][col] : 0.f;
                #pragma unroll
                for (int r = 0; r < 8; ++r) {
                    int row = mrow + r + ((lane >> 4) << 3);
                    _Float16 hv = (_Float16)(acc[r] + bb);
                    if      (mat == 0) s.qs[row][col] = hv;
                    else if (mat == 1) s.ks[row][col] = hv;
                    else               s.vt[col][row] = hv;   // V stored transposed
                }
            }
        }
    }
    __syncthreads();   // xs dead from here -> logits may overlay

    // ---- Phase 3: logits = Q @ K^T   ([128,32] x [32,128]) ----
    for (int jt = 0; jt < 8; ++jt) {
        v16h a  = load_a(&s.qs[0][0], 40, mrow, 0, lane);
        v16h bf = load_b(&s.ks[0][0], 40, jt * 16, 0, lane);  // K is [j][d] == [N][K]
        v8f acc = {};
        acc = __builtin_amdgcn_wmma_f32_16x16x32_f16(
            false, a, false, bf, (short)0, acc, false, false);
        int col = (jt << 4) | (lane & 15);
        #pragma unroll
        for (int r = 0; r < 8; ++r)
            s.ov.logits[mrow + r + ((lane >> 4) << 3)][col] = acc[r];
    }
    __syncthreads();

    // ---- Softmax along j (no scaling, per reference); one thread per row ----
    if (tid < 128) {
        float m = -1e30f;
        for (int j = 0; j < 128; ++j) m = fmaxf(m, s.ov.logits[tid][j]);
        float sum = 0.f;
        for (int j = 0; j < 128; ++j) {
            float e = __expf(s.ov.logits[tid][j] - m);
            s.ov.logits[tid][j] = e;
            sum += e;
        }
        float inv = 1.f / sum;
        for (int j = 0; j < 128; ++j)
            s.probs[tid][j] = (_Float16)(s.ov.logits[tid][j] * inv);
    }
    __syncthreads();

    // ---- Phase 4: ctx = probs @ V  ([128,128] x [128,32]) -> reuse qs ----
    for (int nt = 0; nt < 2; ++nt) {
        v8f acc = {};
        #pragma unroll
        for (int kst = 0; kst < 4; ++kst) {
            v16h a  = load_a(&s.probs[0][0], 136, mrow, kst * 32, lane);
            v16h bf = load_b(&s.vt[0][0], 136, nt * 16, kst * 32, lane); // V^T = [N][K]
            acc = __builtin_amdgcn_wmma_f32_16x16x32_f16(
                false, a, false, bf, (short)0, acc, false, false);
        }
        int col = (nt << 4) | (lane & 15);
        #pragma unroll
        for (int r = 0; r < 8; ++r)
            s.qs[mrow + r + ((lane >> 4) << 3)][col] = (_Float16)acc[r];
    }
    __syncthreads();

    // ---- Phase 5: out = ctx @ Wp + bp ; store transposed to [B,C,H,W] ----
    float* ob = out + ((size_t)(b * EMBED) * HDIM + h) * WDIM;
    for (int ct = 0; ct < 16; ++ct) {          // 16 c-tiles of 16 channels
        {
            v16h a  = load_a(&s.qs[0][0], 40, mrow, 0, lane);
            v16h bf = load_b(wpT, 32, ct * 16, 0, lane);      // Wp^T = [c][d] = [N][K]
            v8f acc = {};
            acc = __builtin_amdgcn_wmma_f32_16x16x32_f16(
                false, a, false, bf, (short)0, acc, false, false);
            int   cl  = lane & 15;
            float bpv = bp_g[ct * 16 + cl];
            #pragma unroll
            for (int r = 0; r < 8; ++r)
                s.outb[cl][mrow + r + ((lane >> 4) << 3)] = acc[r] + bpv;
        }
        __syncthreads();
        for (int i = tid; i < 16 * 128; i += 256) {   // coalesced 512B rows
            int cl = i >> 7, w = i & 127;
            ob[(size_t)(ct * 16 + cl) * (HDIM * WDIM) + w] = s.outb[cl][w];
        }
        __syncthreads();
    }
}

extern "C" void kernel_launch(void* const* d_in, const int* in_sizes, int n_in,
                              void* d_out, int out_size, void* d_ws, size_t ws_size,
                              hipStream_t stream) {
    const float* x  = (const float*)d_in[0];
    const float* Wq = (const float*)d_in[1];
    const float* bq = (const float*)d_in[2];
    const float* Wk = (const float*)d_in[3];
    const float* bk = (const float*)d_in[4];
    const float* Wv = (const float*)d_in[5];
    const float* bv = (const float*)d_in[6];
    const float* Wp = (const float*)d_in[7];
    const float* bp = (const float*)d_in[8];
    _Float16* wts = (_Float16*)d_ws;   // 64 KB of f16 transposed/padded weights

    prep_weights<<<32, 256, 0, stream>>>(Wq, Wk, Wv, Wp, wts);
    attn_row_kernel<<<16 * HDIM, 256, 0, stream>>>(x, bq, bk, bv, bp, wts,
                                                   (float*)d_out);
}
